// NeuralOperator_30494267802188
// MI455X (gfx1250) — matrix operand order repaired and checked by reading, not compiled, not run on hardware
//
#include <hip/hip_runtime.h>
#include <hip/hip_bf16.h>
#include <math.h>
#include <stdint.h>

typedef __attribute__((ext_vector_type(16))) _Float16 v16h;
typedef __attribute__((ext_vector_type(8)))  _Float16 v8h;
typedef __attribute__((ext_vector_type(8)))  float    v8f;

#define PI_F 3.14159265358979323846f

// problem constants
#define BB 4
#define HH 64
#define WW 64
#define CC 32
#define NN 4096           // H*W
#define EE 131072
#define KMID 64

// workspace layout (float offsets)
#define OFF_XHAT   0u
#define OFF_X1     524288u
#define OFF_OUT1   1048576u
#define OFF_Y2     1572864u
#define OFF_AGG    2097152u
#define OFF_URE    2621440u
#define OFF_UIM    2883584u
#define OFF_TRE    3145728u
#define OFF_TIM    3276800u
#define OFF_ARE    3407872u
#define OFF_AIM    3473408u
#define OFF_FRE    3538944u
#define OFF_FIM    3604480u
#define OFF_STX    3670016u   // 512 (mean,rstd) per (b,h)
#define OFF_STX1   3670528u
#define OFF_STY2   3671040u
#define OFF_KW2T   3671552u   // transposed half [1024 cols][64 k] = 65536 halves

__device__ __forceinline__ float gelu_exact(float x) {
    return 0.5f * x * (1.0f + erff(x * 0.70710678118654752f));
}

// ---- InstanceNorm stats over (W,C) = 2048 contiguous floats per (b,h) ----
__global__ void k_stats(const float* __restrict__ x, float* __restrict__ st) {
    int bh = blockIdx.x, t = threadIdx.x;
    const float* p = x + (size_t)bh * 2048;
    float s = 0.f, s2 = 0.f;
    for (int i = t; i < 2048; i += 256) { float v = p[i]; s += v; s2 += v * v; }
    __shared__ float rs[256], rq[256];
    rs[t] = s; rq[t] = s2; __syncthreads();
    for (int o = 128; o > 0; o >>= 1) {
        if (t < o) { rs[t] += rs[t + o]; rq[t] += rq[t + o]; }
        __syncthreads();
    }
    if (t == 0) {
        float m = rs[0] * (1.0f / 2048.0f);
        float v = rq[0] * (1.0f / 2048.0f) - m * m;
        st[2 * bh] = m; st[2 * bh + 1] = rsqrtf(v + 1e-5f);
    }
}

__global__ void k_apply(const float* __restrict__ x, const float* __restrict__ st,
                        float* __restrict__ y) {
    int g = blockIdx.x * 256 + threadIdx.x;
    int bh = g >> 11;
    y[g] = (x[g] - st[2 * bh]) * st[2 * bh + 1];
}

// transpose ker_w2 [64][1024] f32 -> [1024][64] f16 (one-time, contiguous B-frags)
__global__ void k_cvt_kw2t(const float* __restrict__ x, _Float16* __restrict__ y) {
    int g = blockIdx.x * 256 + threadIdx.x;   // 65536
    int col = g >> 6, k = g & 63;
    y[g] = (_Float16)x[k * 1024 + col];
}

// ---- truncated forward DFT along y: T[b,x,ky,c], ky=0..15 ----
__global__ void k_dft_y(const float* __restrict__ xh, float* __restrict__ Tre,
                        float* __restrict__ Tim) {
    int g = blockIdx.x * 256 + threadIdx.x;           // 131072
    int c = g & 31, ky = (g >> 5) & 15, x = (g >> 9) & 63, b = g >> 15;
    const float* p = xh + (size_t)((b * 64 + x) * 64) * 32 + c;
    float sr = 0.f, si = 0.f;
    for (int y = 0; y < 64; y++) {
        float ang = -(PI_F / 32.0f) * (float)((ky * y) & 63);
        float sn, cs; __sincosf(ang, &sn, &cs);
        float v = p[y * 32];
        sr += v * cs; si += v * sn;
    }
    Tre[g] = sr; Tim[g] = si;
}

// ---- along x at the 32 retained kx modes; fold 1/sqrt(H*W)=1/64 ----
__global__ void k_dft_x(const float* __restrict__ Tre, const float* __restrict__ Tim,
                        float* __restrict__ Are, float* __restrict__ Aim) {
    int g = blockIdx.x * 256 + threadIdx.x;           // 65536
    int c = g & 31, ky = (g >> 5) & 15, kxi = (g >> 9) & 31, b = g >> 14;
    int kx = (kxi < 16) ? kxi : kxi + 32;
    float ar = 0.f, ai = 0.f;
    for (int x = 0; x < 64; x++) {
        int idx = ((b * 64 + x) * 16 + ky) * 32 + c;
        float tr = Tre[idx], ti = Tim[idx];
        float ang = -(PI_F / 32.0f) * (float)((kx * x) & 63);
        float sn, cs; __sincosf(ang, &sn, &cs);
        ar += tr * cs - ti * sn;
        ai += tr * sn + ti * cs;
    }
    Are[g] = ar * (1.0f / 64.0f); Aim[g] = ai * (1.0f / 64.0f);
}

// ---- complex channel mix per mode ----
__global__ void k_modemix(const float* __restrict__ Are, const float* __restrict__ Aim,
                          const float* __restrict__ w1r, const float* __restrict__ w1i,
                          const float* __restrict__ w2r, const float* __restrict__ w2i,
                          float* __restrict__ Fre, float* __restrict__ Fim) {
    int g = blockIdx.x * 256 + threadIdx.x;           // 65536
    int o = g & 31, ky = (g >> 5) & 15, kxi = (g >> 9) & 31, b = g >> 14;
    const float* wr = (kxi < 16) ? w1r : w2r;
    const float* wi = (kxi < 16) ? w1i : w2i;
    int kxr = (kxi < 16) ? kxi : kxi - 16;
    float orr = 0.f, oii = 0.f;
    for (int i = 0; i < 32; i++) {
        int aidx = ((b * 32 + kxi) * 16 + ky) * 32 + i;
        float ar = Are[aidx], ai = Aim[aidx];
        int widx = ((i * 32 + o) * 16 + kxr) * 16 + ky;
        float r = wr[widx], im = wi[widx];
        orr += ar * r - ai * im;
        oii += ar * im + ai * r;
    }
    Fre[g] = orr; Fim[g] = oii;
}

// ---- inverse along y (half-spectrum real trick, multiplicity 2 for ky>0) ----
__global__ void k_idft_y(const float* __restrict__ Fre, const float* __restrict__ Fim,
                         float* __restrict__ Ure, float* __restrict__ Uim) {
    int g = blockIdx.x * 256 + threadIdx.x;           // 262144
    int o = g & 31, y = (g >> 5) & 63, kxi = (g >> 11) & 31, b = g >> 16;
    float ur = 0.f, ui = 0.f;
    for (int ky = 0; ky < 16; ky++) {
        int idx = ((b * 32 + kxi) * 16 + ky) * 32 + o;
        float fr = Fre[idx], fi = Fim[idx];
        float ang = (PI_F / 32.0f) * (float)((ky * y) & 63);
        float sn, cs; __sincosf(ang, &sn, &cs);
        float mlt = ky ? 2.0f : 1.0f;
        ur += mlt * (fr * cs - fi * sn);
        ui += mlt * (fr * sn + fi * cs);
    }
    Ure[g] = ur; Uim[g] = ui;
}

// ---- inverse along x, keep real part; fold 1/64 ortho ----
__global__ void k_idft_x(const float* __restrict__ Ure, const float* __restrict__ Uim,
                         float* __restrict__ x1) {
    int g = blockIdx.x * 256 + threadIdx.x;           // 524288
    int o = g & 31, y = (g >> 5) & 63, x = (g >> 11) & 63, b = g >> 17;
    float acc = 0.f;
    for (int kxi = 0; kxi < 32; kxi++) {
        int kx = (kxi < 16) ? kxi : kxi + 32;
        int idx = ((b * 32 + kxi) * 64 + y) * 32 + o;
        float ang = (PI_F / 32.0f) * (float)((kx * x) & 63);
        float sn, cs; __sincosf(ang, &sn, &cs);
        acc += Ure[idx] * cs - Uim[idx] * sn;
    }
    x1[g] = acc * (1.0f / 64.0f);
}

__device__ __forceinline__ v16h cat8(v8h lo, v8h hi) {
    v16h r;
    #pragma unroll
    for (int i = 0; i < 8; i++) { r[i] = lo[i]; r[8 + i] = hi[i]; }
    return r;
}

// ---- pointwise MLP 32->64(gelu)->32 via WMMA f16; optional pre-norm ----
// block = 128 threads = 4 waves, each wave owns one 16-row tile (64 rows/block)
// LDS weights stored TRANSPOSED [col][K] so each B fragment is one aligned v16h.
__global__ void k_mlp(const float* __restrict__ in, const float* __restrict__ stats,
                      const float* __restrict__ w1, const float* __restrict__ bb1,
                      const float* __restrict__ w2, const float* __restrict__ bb2,
                      float* __restrict__ out) {
    __shared__ _Float16 w1t[2048];        // [64 cols][32 k]
    __shared__ _Float16 w2t[2048];        // [32 cols][64 k]
    __shared__ _Float16 hbuf[4][1024];    // per-wave [16 rows][64]
    int t = threadIdx.x;
    for (int i = t; i < 2048; i += 128) {
        w1t[i] = (_Float16)w1[(i & 31) * 64 + (i >> 5)];
        w2t[i] = (_Float16)w2[(i & 63) * 32 + (i >> 6)];
    }
    __syncthreads();
    int wv = t >> 5, lane = t & 31, m = lane & 15, half = lane >> 4;
    int row0 = blockIdx.x * 64 + wv * 16;
    int row = row0 + m;
    float mean = 0.f, rstd = 1.f;
    if (stats) { int bh = row >> 6; mean = stats[2 * bh]; rstd = stats[2 * bh + 1]; }
    const float* rp = in + (size_t)row * 32;
    v16h a;                                // A 16x32: lane m=row, K contiguous = 16*half + i
    #pragma unroll
    for (int i = 0; i < 16; i++) {
        int K = 16 * half + i;
        float v = rp[K];
        if (stats) v = (v - mean) * rstd;
        a[i] = (_Float16)v;
    }
    #pragma unroll
    for (int nt = 0; nt < 4; nt++) {       // layer 1: N=64 in 4 tiles
        int col = nt * 16 + m;
        v16h bf = *(const v16h*)&w1t[col * 32 + 16 * half];
        float bias = bb1[col];
        v8f c;
        #pragma unroll
        for (int j = 0; j < 8; j++) c[j] = bias;
        v8f d = __builtin_amdgcn_wmma_f32_16x16x32_f16(false, a, false, bf, (short)0, c, false, false);
        #pragma unroll
        for (int j = 0; j < 8; j++)
            hbuf[wv][(j + 8 * half) * 64 + col] = (_Float16)gelu_exact(d[j]);
    }
    __syncthreads();
    // layer 2 A fragments: two contiguous 8-half chunks per register half
    v16h a0 = cat8(*(const v8h*)&hbuf[wv][m * 64 + 16 * half],
                   *(const v8h*)&hbuf[wv][m * 64 + 16 * half + 8]);
    v16h a1 = cat8(*(const v8h*)&hbuf[wv][m * 64 + 32 + 16 * half],
                   *(const v8h*)&hbuf[wv][m * 64 + 32 + 16 * half + 8]);
    #pragma unroll
    for (int nt = 0; nt < 2; nt++) {
        int col = nt * 16 + m;
        v16h b0 = *(const v16h*)&w2t[col * 64 + 16 * half];
        v16h b1 = *(const v16h*)&w2t[col * 64 + 32 + 16 * half];
        float bias = bb2[col];
        v8f c;
        #pragma unroll
        for (int j = 0; j < 8; j++) c[j] = bias;
        v8f d = __builtin_amdgcn_wmma_f32_16x16x32_f16(false, a0, false, b0, (short)0, c, false, false);
        d     = __builtin_amdgcn_wmma_f32_16x16x32_f16(false, a1, false, b1, (short)0, d, false, false);
        #pragma unroll
        for (int j = 0; j < 8; j++)
            out[(size_t)(row0 + j + 8 * half) * 32 + col] = d[j];
    }
}

// ---- fused GNO: edge MLP layer2 via WMMA + per-edge GEMV + atomic scatter ----
// block = 128 threads = 4 waves; kw2 (transposed f16, 128KB) staged into LDS with
// CDNA5 async direct-to-LDS loads (no VGPR round-trip), grid-stride over tiles.
#define GNO_BLOCKS 256
__global__ void k_gno(const float* __restrict__ xhat, const int* __restrict__ eidx,
                      const float* __restrict__ eattr,
                      const float* __restrict__ kw1, const float* __restrict__ kb1,
                      const _Float16* __restrict__ kw2t, const float* __restrict__ kb2,
                      float* __restrict__ agg) {
    __shared__ _Float16 kw2s[65536];      // [1024 cols][64 k] transposed, 128KB
    __shared__ _Float16 xs[4][2048];      // per-wave [32 ci][16 e][4 b] f16, 16KB
    int t = threadIdx.x, wv = t >> 5, lane = t & 31, m = lane & 15, half = lane >> 4;

    // async stage: 8192 x 16B chunks, per-lane GLOBAL_LOAD_ASYNC_TO_LDS_B128
    {
        uint32_t lbase = (uint32_t)(size_t)&kw2s[0];        // LDS byte offset
        const char* gb = (const char*)kw2t;
        for (int i = t; i < 8192; i += 128) {
            uint32_t loff = lbase + (uint32_t)i * 16u;
            uint64_t ga = (uint64_t)(size_t)(gb + (size_t)i * 16u);
            asm volatile("global_load_async_to_lds_b128 %0, %1, off"
                         :: "v"(loff), "v"(ga) : "memory");
        }
        asm volatile("s_wait_asynccnt 0" ::: "memory");     // drain this wave's ASYNCcnt
    }
    __syncthreads();                                        // cross-wave visibility

    for (int tile = blockIdx.x * 4 + wv; tile < EE / 16; tile += GNO_BLOCKS * 4) {
        int e0 = tile * 16;
        // prefetch next tile's edge data (addresses known one iteration ahead)
        if (tile + GNO_BLOCKS * 4 < EE / 16) {
            int en = e0 + GNO_BLOCKS * 4 * 16;
            __builtin_prefetch(&eattr[(size_t)en * 6 + lane], 0, 3);
            __builtin_prefetch(&eidx[en], 0, 3);
            __builtin_prefetch(&eidx[EE + en], 0, 3);
        }
        // gather normalized src features, layout [ci][edge][b]
        for (int i = lane; i < 2048; i += 32) {
            int c = i >> 6, el = (i >> 2) & 15, b = i & 3;
            int s = eidx[e0 + el];
            xs[wv][i] = (_Float16)xhat[(size_t)(b * NN + s) * 32 + c];
        }
        // layer 1 directly into this lane's fragment slots (edge m, K = 16*half + i)
        float at[6];
        #pragma unroll
        for (int d = 0; d < 6; d++) at[d] = eattr[(size_t)(e0 + m) * 6 + d];
        v16h a0, a1;
        #pragma unroll
        for (int i = 0; i < 16; i++) {
            int K = 16 * half + i;
            float v0 = kb1[K], v1 = kb1[K + 32];
            #pragma unroll
            for (int d = 0; d < 6; d++) {
                v0 += at[d] * kw1[d * 64 + K];
                v1 += at[d] * kw1[d * 64 + K + 32];
            }
            a0[i] = (_Float16)gelu_exact(v0);
            a1[i] = (_Float16)gelu_exact(v1);
        }
        __syncthreads();

        float mA[32], mB[32];
        #pragma unroll
        for (int i = 0; i < 32; i++) { mA[i] = 0.f; mB[i] = 0.f; }

        for (int nt2 = 0; nt2 < 32; nt2++) {
            int colA = nt2 * 32 + m;          // co = m        (ci = nt2)
            int colB = colA + 16;             // co = 16 + m
            v16h bA0 = *(const v16h*)&kw2s[colA * 64 + 16 * half];
            v16h bA1 = *(const v16h*)&kw2s[colA * 64 + 32 + 16 * half];
            v16h bB0 = *(const v16h*)&kw2s[colB * 64 + 16 * half];
            v16h bB1 = *(const v16h*)&kw2s[colB * 64 + 32 + 16 * half];
            v8f cA, cB;
            float biasA = kb2[colA], biasB = kb2[colB];
            #pragma unroll
            for (int j = 0; j < 8; j++) { cA[j] = biasA; cB[j] = biasB; }
            v8f dA = __builtin_amdgcn_wmma_f32_16x16x32_f16(false, a0, false, bA0, (short)0, cA, false, false);
            dA     = __builtin_amdgcn_wmma_f32_16x16x32_f16(false, a1, false, bA1, (short)0, dA, false, false);
            v8f dB = __builtin_amdgcn_wmma_f32_16x16x32_f16(false, a0, false, bB0, (short)0, cB, false, false);
            dB     = __builtin_amdgcn_wmma_f32_16x16x32_f16(false, a1, false, bB1, (short)0, dB, false, false);
            // src features for ci = nt2, this lane's 8 edge rows, all 4 batches
            v16h xv0 = *(const v16h*)&xs[wv][nt2 * 64 + 32 * half];
            v16h xv1 = *(const v16h*)&xs[wv][nt2 * 64 + 32 * half + 16];
            #pragma unroll
            for (int j = 0; j < 8; j++) {
                #pragma unroll
                for (int b = 0; b < 4; b++) {
                    float xf = (float)((j < 4) ? xv0[j * 4 + b] : xv1[(j - 4) * 4 + b]);
                    mA[j * 4 + b] += xf * dA[j];
                    mB[j * 4 + b] += xf * dB[j];
                }
            }
        }
        // direct scatter: each lane owns disjoint (edge-row set, co) slots
        #pragma unroll
        for (int j = 0; j < 8; j++) {
            int er = j + 8 * half;
            int dd = eidx[EE + e0 + er];
            #pragma unroll
            for (int b = 0; b < 4; b++) {
                atomicAdd(&agg[(size_t)(b * NN + dd) * 32 + m],      mA[j * 4 + b]);
                atomicAdd(&agg[(size_t)(b * NN + dd) * 32 + 16 + m], mB[j * 4 + b]);
            }
        }
        __syncthreads();
    }
}

// ---- final: gelu(out1 + inorm(y2) + agg + xhat@root + gbias) ----
__global__ void k_final(const float* __restrict__ out1, const float* __restrict__ y2,
                        const float* __restrict__ st2, const float* __restrict__ agg,
                        const float* __restrict__ xhat, const float* __restrict__ root,
                        const float* __restrict__ gbias, float* __restrict__ out) {
    int g = blockIdx.x * 256 + threadIdx.x;           // 524288
    int c = g & 31;
    int bh = g >> 11;
    float m2 = st2[2 * bh], r2 = st2[2 * bh + 1];
    const float* xr = xhat + (size_t)(g >> 5) * 32;
    float acc = gbias[c];
    #pragma unroll
    for (int i = 0; i < 32; i++) acc += xr[i] * root[i * 32 + c];
    float v = out1[g] + (y2[g] - m2) * r2 + agg[g] + acc;
    out[g] = gelu_exact(v);
}

extern "C" void kernel_launch(void* const* d_in, const int* in_sizes, int n_in,
                              void* d_out, int out_size, void* d_ws, size_t ws_size,
                              hipStream_t stream) {
    const float* nodes   = (const float*)d_in[0];
    const int*   eidx    = (const int*)  d_in[1];
    const float* eattr   = (const float*)d_in[2];
    const float* w1r     = (const float*)d_in[3];
    const float* w1i     = (const float*)d_in[4];
    const float* w2r     = (const float*)d_in[5];
    const float* w2i     = (const float*)d_in[6];
    const float* mlp_w1  = (const float*)d_in[7];
    const float* mlp_b1  = (const float*)d_in[8];
    const float* mlp_w2  = (const float*)d_in[9];
    const float* mlp_b2  = (const float*)d_in[10];
    const float* wm_w1   = (const float*)d_in[11];
    const float* wm_b1   = (const float*)d_in[12];
    const float* wm_w2   = (const float*)d_in[13];
    const float* wm_b2   = (const float*)d_in[14];
    const float* ker_w1  = (const float*)d_in[15];
    const float* ker_b1  = (const float*)d_in[16];
    const float* ker_w2  = (const float*)d_in[17];
    const float* ker_b2  = (const float*)d_in[18];
    const float* root    = (const float*)d_in[19];
    const float* gbias   = (const float*)d_in[20];

    float* w = (float*)d_ws;
    float* xhat = w + OFF_XHAT;  float* x1   = w + OFF_X1;
    float* out1 = w + OFF_OUT1;  float* y2   = w + OFF_Y2;
    float* agg  = w + OFF_AGG;
    float* Ure  = w + OFF_URE;   float* Uim  = w + OFF_UIM;
    float* Tre  = w + OFF_TRE;   float* Tim  = w + OFF_TIM;
    float* Are  = w + OFF_ARE;   float* Aim  = w + OFF_AIM;
    float* Fre  = w + OFF_FRE;   float* Fim  = w + OFF_FIM;
    float* stx  = w + OFF_STX;   float* stx1 = w + OFF_STX1;
    float* sty2 = w + OFF_STY2;
    _Float16* kw2t = (_Float16*)(w + OFF_KW2T);
    float* out = (float*)d_out;

    hipMemsetAsync(agg, 0, (size_t)BB * NN * CC * sizeof(float), stream);

    // normalize input + transposed f16 weight conversion
    k_stats<<<256, 256, 0, stream>>>(nodes, stx);
    k_apply<<<2048, 256, 0, stream>>>(nodes, stx, xhat);
    k_cvt_kw2t<<<256, 256, 0, stream>>>(ker_w2, kw2t);

    // spectral branch (truncated-mode separable DFT)
    k_dft_y<<<512, 256, 0, stream>>>(xhat, Tre, Tim);
    k_dft_x<<<256, 256, 0, stream>>>(Tre, Tim, Are, Aim);
    k_modemix<<<256, 256, 0, stream>>>(Are, Aim, w1r, w1i, w2r, w2i, Fre, Fim);
    k_idft_y<<<1024, 256, 0, stream>>>(Fre, Fim, Ure, Uim);
    k_idft_x<<<2048, 256, 0, stream>>>(Ure, Uim, x1);
    k_stats<<<256, 256, 0, stream>>>(x1, stx1);
    k_mlp<<<256, 128, 0, stream>>>(x1, stx1, mlp_w1, mlp_b1, mlp_w2, mlp_b2, out1);

    // pointwise branch (norm applied after, inside k_final)
    k_mlp<<<256, 128, 0, stream>>>(xhat, nullptr, wm_w1, wm_b1, wm_w2, wm_b2, y2);
    k_stats<<<256, 256, 0, stream>>>(y2, sty2);

    // GNO branch: fused kernel-MLP + message + scatter
    k_gno<<<GNO_BLOCKS, 128, 0, stream>>>(xhat, eidx, eattr, ker_w1, ker_b1, kw2t, ker_b2, agg);

    // combine
    k_final<<<2048, 256, 0, stream>>>(out1, y2, sty2, agg, xhat, root, gbias, out);
}
